// VNCrossAttention_28982439313690
// MI455X (gfx1250) — compile-verified, matrix operand order checked by reading, not run on hardware
//
#include <hip/hip_runtime.h>

typedef __attribute__((ext_vector_type(16))) _Float16 v16h;
typedef __attribute__((ext_vector_type(8)))  _Float16 v8h;
typedef __attribute__((ext_vector_type(8)))  float    v8f;

#define BATCH 4
#define NSEQ  2048
#define HEADS 8
#define DHEAD 16
#define CIN   128
#define EREAL 48
#define EP    64
#define QTILE 128
#define KTILE 64
#define TOK_TILE 64
#define BIAS_EPS 1e-6f

// ---------------------------------------------------------------------------
// WMMA helpers (gfx1250 wave32, 16x16x32 f16 -> f32)
// ---------------------------------------------------------------------------
__device__ __forceinline__ v8f wmma16(v16h a, v16h b, v8f c) {
  return __builtin_amdgcn_wmma_f32_16x16x32_f16(false, a, false, b, (short)0, c,
                                                false, false);
}

// Load a 16x32 f16 A/B fragment from a row-major f16 array.
// Assumed layout (per CDNA5 ISA 16-bit A 16x32):
//   lanes 0-15  : elems 0-7 -> k0+0..7,  elems 8-15 -> k0+16..23
//   lanes 16-31 : elems 0-7 -> k0+8..15, elems 8-15 -> k0+24..31
__device__ __forceinline__ v16h load_frag16(const _Float16* base, int row,
                                            int stride, int k0, int lane) {
  const int hi = (lane & 16) ? 8 : 0;
  const _Float16* p = base + (size_t)row * stride + k0 + hi;
  v8h lo = *(const v8h*)(p);
  v8h hh = *(const v8h*)(p + 16);
  v16h r;
#pragma unroll
  for (int j = 0; j < 8; ++j) { r[j] = lo[j]; r[8 + j] = hh[j]; }
  return r;
}

// A fragment of W (f32 row-major 128x128), converted f32 -> f16 in flight.
__device__ __forceinline__ v16h load_frag_w(const float* W, int row, int k0,
                                            int lane) {
  const int hi = (lane & 16) ? 8 : 0;
  const float* p = W + row * CIN + k0 + hi;
  v16h r;
#pragma unroll
  for (int j = 0; j < 8; ++j) {
    r[j]     = (_Float16)p[j];
    r[8 + j] = (_Float16)p[16 + j];
  }
  return r;
}

// Async DMA: 16 bytes global -> LDS, tracked by ASYNCcnt (CDNA5).
__device__ __forceinline__ void async_copy_b128(unsigned lds_off,
                                                const void* gptr) {
  unsigned long long ga = (unsigned long long)(uintptr_t)gptr;
  asm volatile("global_load_async_to_lds_b128 %0, %1, off"
               :: "v"(lds_off), "v"(ga) : "memory");
}

__device__ __forceinline__ unsigned lds_addr32(const void* p) {
  // Generic (flat) LDS pointer carries the group-segment byte offset in
  // addr[31:0] (aperture in the high half) -> truncate.
  return (unsigned)(uintptr_t)p;
}

// ---------------------------------------------------------------------------
// Rotation-equivariant projection: Y = W X (per token 128x3), then
// out = Y * (1 + b/(||Y||_coord + eps)).
// mode 0: f16 head layout [b][h][n][EP]    (e = d*3+c, pad e>=48 zeroed)
// mode 2: f16 head layout [b][h][EP][n]    (transposed, for V)
// mode 1: f32 [tok][128][3]
// ---------------------------------------------------------------------------
__global__ __launch_bounds__(256) void vnproj_kernel(
    const float* __restrict__ X, const float* __restrict__ W,
    const float* __restrict__ bias, _Float16* __restrict__ outH,
    float* __restrict__ outF, int mode) {
  __shared__ _Float16 Xlds[3 * TOK_TILE * CIN];  // [c][tok][i], 48 KB

  const int tid  = threadIdx.x;
  const int lane = tid & 31;
  const int wave = tid >> 5;
  const int t0   = blockIdx.x * TOK_TILE;

  // Cooperative load+transpose: global [tok][i][c] f32 -> LDS [c][tok][i] f16
  const float4* in4 = (const float4*)(X + (size_t)t0 * CIN * 3);
  for (int vi = tid; vi < TOK_TILE * CIN * 3 / 4; vi += 256) {
    float4 dv = in4[vi];
    int f0 = vi * 4;
#pragma unroll
    for (int j = 0; j < 4; ++j) {
      int f   = f0 + j;
      int tok = f / (CIN * 3);
      int rem = f - tok * (CIN * 3);
      int i   = rem / 3;
      int c   = rem - i * 3;
      float val = (j == 0) ? dv.x : (j == 1) ? dv.y : (j == 2) ? dv.z : dv.w;
      Xlds[(c * TOK_TILE + tok) * CIN + i] = (_Float16)val;
    }
  }
  __syncthreads();

  v8f acc[3][4];
#pragma unroll
  for (int c = 0; c < 3; ++c)
#pragma unroll
    for (int nf = 0; nf < 4; ++nf)
#pragma unroll
      for (int j = 0; j < 8; ++j) acc[c][nf][j] = 0.f;

  const int orow = wave * 16 + (lane & 15);
#pragma unroll
  for (int kc = 0; kc < 4; ++kc) {
    v16h a = load_frag_w(W, orow, kc * 32, lane);
#pragma unroll
    for (int c = 0; c < 3; ++c)
#pragma unroll
      for (int nf = 0; nf < 4; ++nf) {
        v16h bfr = load_frag16(Xlds + c * TOK_TILE * CIN,
                               nf * 16 + (lane & 15), CIN, kc * 32, lane);
        acc[c][nf] = wmma16(a, bfr, acc[c][nf]);
      }
  }

  // Epilogue: all 3 coords of (o, tok) live in the same lane/VGPR slot.
  const int hi8 = (lane & 16) ? 8 : 0;
#pragma unroll
  for (int nf = 0; nf < 4; ++nf) {
    int tok = t0 + nf * 16 + (lane & 15);
    int bb  = tok / NSEQ;
    int n   = tok - bb * NSEQ;
#pragma unroll
    for (int r = 0; r < 8; ++r) {
      int o = wave * 16 + r + hi8;
      float y0 = acc[0][nf][r], y1 = acc[1][nf][r], y2 = acc[2][nf][r];
      float nrm = sqrtf(y0 * y0 + y1 * y1 + y2 * y2);
      float fct = 1.0f + bias[o] / (nrm + BIAS_EPS);
      if (mode == 0) {
        int hh = o >> 4, dd = o & 15;
        size_t base = ((size_t)(bb * HEADS + hh) * NSEQ + n) * EP + dd * 3;
        outH[base + 0] = (_Float16)(y0 * fct);
        outH[base + 1] = (_Float16)(y1 * fct);
        outH[base + 2] = (_Float16)(y2 * fct);
      } else if (mode == 2) {
        int hh = o >> 4, dd = o & 15;
        size_t base = ((size_t)(bb * HEADS + hh) * EP + dd * 3) * NSEQ + n;
        outH[base]            = (_Float16)(y0 * fct);
        outH[base + NSEQ]     = (_Float16)(y1 * fct);
        outH[base + 2 * NSEQ] = (_Float16)(y2 * fct);
      } else {
        size_t base = ((size_t)tok * CIN + o) * 3;
        outF[base + 0] = y0 * fct;
        outF[base + 1] = y1 * fct;
        outF[base + 2] = y2 * fct;
      }
    }
  }
  if (mode == 0) {  // zero the e = 48..63 pad
    for (int z = tid; z < TOK_TILE * HEADS * (EP - EREAL); z += 256) {
      int tok = t0 + z / (HEADS * (EP - EREAL));
      int rem = z % (HEADS * (EP - EREAL));
      int hh  = rem / (EP - EREAL);
      int e   = EREAL + rem % (EP - EREAL);
      int bb = tok / NSEQ, n = tok % NSEQ;
      outH[((size_t)(bb * HEADS + hh) * NSEQ + n) * EP + e] = (_Float16)0.f;
    }
  } else if (mode == 2) {
    for (int z = tid; z < TOK_TILE * HEADS * (EP - EREAL); z += 256) {
      int tok = t0 + z / (HEADS * (EP - EREAL));
      int rem = z % (HEADS * (EP - EREAL));
      int hh  = rem / (EP - EREAL);
      int e   = EREAL + rem % (EP - EREAL);
      int bb = tok / NSEQ, n = tok % NSEQ;
      outH[((size_t)(bb * HEADS + hh) * EP + e) * NSEQ + n] = (_Float16)0.f;
    }
  }
}

// ---------------------------------------------------------------------------
// Flash-style attention: one workgroup per (b, h, 128-row Q tile).
// 8 waves x 16 Q rows; K/V double-buffered in LDS via async DMA; online
// softmax; V workspace is pre-transposed so the DMA is a plain 2D copy.
// ---------------------------------------------------------------------------
__global__ __launch_bounds__(256) void vnattn_kernel(
    const _Float16* __restrict__ Qh, const _Float16* __restrict__ Kh,
    const _Float16* __restrict__ Vt, float* __restrict__ Xmid) {
  __shared__ _Float16 Kbuf[2][KTILE * EP];   // [krow][e]        2 x 8 KB
  __shared__ _Float16 Vbuf[2][EP * KTILE];   // [e][krow]        2 x 8 KB
  __shared__ _Float16 Ptile[8 * 16 * EP];    // per-wave P          16 KB

  const int tid = threadIdx.x, lane = tid & 31, wave = tid >> 5;
  const int nqt = NSEQ / QTILE;
  const int qt  = blockIdx.x % nqt;
  const int bh  = blockIdx.x / nqt;
  const int h   = bh % HEADS, b = bh / HEADS;

  const _Float16* Qb = Qh + (size_t)bh * NSEQ * EP;
  const _Float16* Kb = Kh + (size_t)bh * NSEQ * EP;   // [n][e]
  const _Float16* Vb = Vt + (size_t)bh * EP * NSEQ;   // [e][n]

  const int qrow0 = qt * QTILE + wave * 16;
  v16h qa0 = load_frag16(Qb, qrow0 + (lane & 15), EP, 0, lane);
  v16h qa1 = load_frag16(Qb, qrow0 + (lane & 15), EP, 32, lane);

  float m[8], l[8];
  v8f o[4];
#pragma unroll
  for (int r = 0; r < 8; ++r) { m[r] = -1e30f; l[r] = 0.f; }
#pragma unroll
  for (int ef = 0; ef < 4; ++ef)
#pragma unroll
    for (int j = 0; j < 8; ++j) o[ef][j] = 0.f;

  const float kscale = 0.14433756729740643f * 1.4426950408889634f; // 48^-.5*log2e
  _Float16* Pw = Ptile + wave * 16 * EP;
  const int hi8 = (lane & 16) ? 8 : 0;
  const int NT = NSEQ / KTILE;

  // Per-thread async issue of one K/V tile pair (4 x b128 per thread).
  auto issue_tile = [&](int kt2, int buf) {
    // K tile: 8 KB contiguous ([64 rows][EP]).
    const char* gk = (const char*)(Kb + (size_t)kt2 * KTILE * EP);
    unsigned kb = lds_addr32(&Kbuf[buf][0]);
    async_copy_b128(kb + tid * 16, gk + tid * 16);
    async_copy_b128(kb + 4096 + tid * 16, gk + 4096 + tid * 16);
    // V tile: 64 rows (e) of 128 B, global row stride NSEQ*2 B.
    unsigned vb = lds_addr32(&Vbuf[buf][0]);
#pragma unroll
    for (int j = 0; j < 2; ++j) {
      int chunk = tid * 2 + j;            // 0..511
      int e   = chunk >> 3;
      int col = (chunk & 7) * 8;          // f16 units
      const _Float16* gv = Vb + (size_t)e * NSEQ + kt2 * KTILE + col;
      async_copy_b128(vb + (e * KTILE + col) * 2, gv);
    }
  };

  issue_tile(0, 0);

  for (int kt = 0; kt < NT; ++kt) {
    const int buf = kt & 1;
    if (kt + 1 < NT) {
      issue_tile(kt + 1, buf ^ 1);
      // Loads retire in order: <=4 outstanding => current tile resident.
      asm volatile("s_wait_asynccnt 4" ::: "memory");
    } else {
      asm volatile("s_wait_asynccnt 0" ::: "memory");
    }
    __syncthreads();

    const _Float16* Ktile = &Kbuf[buf][0];
    const _Float16* Vtile = &Vbuf[buf][0];

    // S = Q K^T  (K-dim = EP = 64, two 32-wide WMMA steps)
    v8f s[4];
#pragma unroll
    for (int nf = 0; nf < 4; ++nf) {
      v8f z;
#pragma unroll
      for (int j = 0; j < 8; ++j) z[j] = 0.f;
      v16h b0 = load_frag16(Ktile, nf * 16 + (lane & 15), EP, 0, lane);
      v16h b1 = load_frag16(Ktile, nf * 16 + (lane & 15), EP, 32, lane);
      z = wmma16(qa0, b0, z);
      z = wmma16(qa1, b1, z);
      s[nf] = z;
    }

    // Online softmax (rows live per 16-lane half -> xor-shuffle reduce)
    float alpha[8];
#pragma unroll
    for (int r = 0; r < 8; ++r) {
      float mx = fmaxf(fmaxf(s[0][r], s[1][r]), fmaxf(s[2][r], s[3][r]));
#pragma unroll
      for (int off = 1; off < 16; off <<= 1)
        mx = fmaxf(mx, __shfl_xor(mx, off, 32));
      float mn = fmaxf(m[r], mx);
      alpha[r] = exp2f((m[r] - mn) * kscale);
      m[r] = mn;
    }
    float ps[8];
#pragma unroll
    for (int r = 0; r < 8; ++r) ps[r] = 0.f;
#pragma unroll
    for (int nf = 0; nf < 4; ++nf)
#pragma unroll
      for (int r = 0; r < 8; ++r) {
        float p = exp2f((s[nf][r] - m[r]) * kscale);
        ps[r] += p;
        Pw[(r + hi8) * EP + nf * 16 + (lane & 15)] = (_Float16)p;
      }
#pragma unroll
    for (int r = 0; r < 8; ++r) {
      float sum = ps[r];
#pragma unroll
      for (int off = 1; off < 16; off <<= 1) sum += __shfl_xor(sum, off, 32);
      l[r] = l[r] * alpha[r] + sum;
    }
#pragma unroll
    for (int ef = 0; ef < 4; ++ef)
#pragma unroll
      for (int r = 0; r < 8; ++r) o[ef][r] *= alpha[r];

    // O += P V  (P re-read from wave-private LDS in A-fragment layout)
    v16h pa0 = load_frag16(Pw, (lane & 15), EP, 0, lane);
    v16h pa1 = load_frag16(Pw, (lane & 15), EP, 32, lane);
#pragma unroll
    for (int ef = 0; ef < 4; ++ef) {
      v16h vb0 = load_frag16(Vtile, ef * 16 + (lane & 15), KTILE, 0, lane);
      v16h vb1 = load_frag16(Vtile, ef * 16 + (lane & 15), KTILE, 32, lane);
      o[ef] = wmma16(pa0, vb0, o[ef]);
      o[ef] = wmma16(pa1, vb1, o[ef]);
    }
    __syncthreads();  // all waves done reading buf before it is re-filled
  }

  // Write merged-head f32 intermediate [tok][128][3], channel = h*16+d
#pragma unroll
  for (int r = 0; r < 8; ++r) {
    float inv = 1.0f / l[r];
    int qrow = qrow0 + r + hi8;
    size_t tokbase = ((size_t)(b * NSEQ + qrow) * CIN) * 3;
#pragma unroll
    for (int ef = 0; ef < 4; ++ef) {
      int e = ef * 16 + (lane & 15);
      if (e < EREAL) {
        int dd = e / 3, c = e - dd * 3;
        Xmid[tokbase + (h * DHEAD + dd) * 3 + c] = o[ef][r] * inv;
      }
    }
  }
}

// ---------------------------------------------------------------------------
extern "C" void kernel_launch(void* const* d_in, const int* in_sizes, int n_in,
                              void* d_out, int out_size, void* d_ws,
                              size_t ws_size, hipStream_t stream) {
  (void)in_sizes; (void)n_in; (void)out_size; (void)ws_size;
  const float* q  = (const float*)d_in[0];
  const float* v  = (const float*)d_in[1];
  const float* Wq = (const float*)d_in[2];
  const float* bq = (const float*)d_in[3];
  const float* Wk = (const float*)d_in[4];
  const float* bk = (const float*)d_in[5];
  const float* Wv = (const float*)d_in[6];
  const float* bv = (const float*)d_in[7];
  const float* Wp = (const float*)d_in[8];
  const float* bp = (const float*)d_in[9];
  float* out = (float*)d_out;

  const size_t headElems = (size_t)BATCH * HEADS * NSEQ * EP;  // f16 each
  _Float16* Qh = (_Float16*)d_ws;
  _Float16* Kh = Qh + headElems;
  _Float16* Vh = Kh + headElems;                 // transposed [b][h][e][n]
  float* Xmid  = (float*)(Vh + headElems);       // [B*N][128][3] f32

  const int tokens = BATCH * NSEQ;
  dim3 blk(256);
  vnproj_kernel<<<tokens / TOK_TILE, blk, 0, stream>>>(q, Wq, bq, Qh, nullptr, 0);
  vnproj_kernel<<<tokens / TOK_TILE, blk, 0, stream>>>(v, Wk, bk, Kh, nullptr, 0);
  vnproj_kernel<<<tokens / TOK_TILE, blk, 0, stream>>>(v, Wv, bv, Vh, nullptr, 2);
  vnattn_kernel<<<BATCH * HEADS * (NSEQ / QTILE), blk, 0, stream>>>(Qh, Kh, Vh,
                                                                    Xmid);
  vnproj_kernel<<<tokens / TOK_TILE, blk, 0, stream>>>(Xmid, Wp, bp, nullptr,
                                                       out, 1);
}
